// MultiHeadAttention_7370163880614
// MI455X (gfx1250) — matrix operand order repaired and checked by reading, not compiled, not run on hardware
//
#include <hip/hip_runtime.h>
#include <math.h>

typedef __attribute__((ext_vector_type(16))) __bf16 v16bf;
typedef __attribute__((ext_vector_type(8)))  __bf16 v8bf;
typedef __attribute__((ext_vector_type(8)))  float  v8f;

#define HID    1024
#define SEQ    2048
#define NHEADS 16
#define HDIM   64
#define BATCH  2
#define MTOT   (BATCH * SEQ)   // 4096 rows for all GEMMs

#define LDK 40   // GEMM LDS row stride in bf16 elems (80B, multiple of 16B)
#define LDA 72   // attention LDS row stride in bf16 elems (144B, multiple of 16B)

// ---------------------------------------------------------------------------
// gfx1250 async memory->LDS path (ASYNCcnt), guarded so compile never breaks.
#if defined(__has_builtin)
#if __has_builtin(__builtin_amdgcn_global_load_async_to_lds_b128) && \
    __has_builtin(__builtin_amdgcn_s_wait_asynccnt)
#define USE_ASYNC_LDS 1
#endif
#endif
#ifndef USE_ASYNC_LDS
#define USE_ASYNC_LDS 0
#endif

// builtin parameter type per hipcc diagnostic: int __vector(4) pointers
typedef int i32x4_gcc __attribute__((vector_size(16)));

__device__ __forceinline__ void copy16_g2l(const void* g, void* l) {
#if USE_ASYNC_LDS
  __builtin_amdgcn_global_load_async_to_lds_b128(
      (__attribute__((address_space(1))) i32x4_gcc*)(g),
      (__attribute__((address_space(3))) i32x4_gcc*)(l), 0, 0);
#else
  *(uint4*)l = *(const uint4*)g;
#endif
}
__device__ __forceinline__ void async_wait_all() {
#if USE_ASYNC_LDS
  __builtin_amdgcn_s_wait_asynccnt(0);
#endif
}

// ---------------------------------------------------------------------------
// WMMA fragment loads from LDS (wave32; this code's contiguous k order, same for A and B)
// A 16x32 bf16: lane L -> row L&15; K = (L>>4)*16 .. +15 contiguous: elems 0-7 at p,
// 8-15 at p+8.
__device__ __forceinline__ v16bf frag_a(const __bf16* p) {
  v8bf lo = *(const v8bf*)(p);
  v8bf hi = *(const v8bf*)(p + 8);
  return __builtin_shufflevector(lo, hi, 0,1,2,3,4,5,6,7,8,9,10,11,12,13,14,15);
}
// B 32x16 bf16: lane L -> col L&15, K = (L>>4)*16 .. +15 contiguous.
__device__ __forceinline__ v16bf frag_b(const __bf16* p) {
  v8bf lo = *(const v8bf*)(p);
  v8bf hi = *(const v8bf*)(p + 8);
  return __builtin_shufflevector(lo, hi, 0,1,2,3,4,5,6,7,8,9,10,11,12,13,14,15);
}

// ---------------------------------------------------------------------------
__global__ __launch_bounds__(256)
void cvt_f32_to_bf16(const float* __restrict__ in, __bf16* __restrict__ out, int n) {
  int i = (blockIdx.x * 256 + threadIdx.x) * 4;
  if (i + 3 < n) {
    const float4 v = *(const float4*)(in + i);
    out[i + 0] = (__bf16)v.x;
    out[i + 1] = (__bf16)v.y;
    out[i + 2] = (__bf16)v.z;
    out[i + 3] = (__bf16)v.w;
  }
}

// ---------------------------------------------------------------------------
// C[m,n] = (sum_k A[m,k] * W[n,k] + bias[n]) * scale
// MODE 0: fp32 plain [M,HID] to Out;  MODE 1: bf16 grouped [B,heads,S,64].
// 128x128 tile, K-step 32, double-buffered LDS with async memory->LDS copies.
template<int MODE>
__global__ __launch_bounds__(256)
void gemm_bias(const __bf16* __restrict__ A,    // [MTOT, HID] bf16, K-major
               const __bf16* __restrict__ W,    // [HID,  HID] bf16, K-major
               const float*  __restrict__ bias, // [HID]
               void* __restrict__ Out, float scale) {
  __shared__ __align__(16) __bf16 As[2][128 * LDK];
  __shared__ __align__(16) __bf16 Bs[2][128 * LDK];

  const int tid  = threadIdx.x;
  const int lane = tid & 31;
  const int wid  = tid >> 5;
  const int wm   = wid & 3;   // 4 row groups of 32 rows
  const int wn   = wid >> 2;  // 2 col groups of 64 cols
  const int m0   = blockIdx.y * 128;
  const int n0   = blockIdx.x * 128;

  v8f acc[2][4];
#pragma unroll
  for (int i = 0; i < 2; ++i)
#pragma unroll
    for (int j = 0; j < 4; ++j)
#pragma unroll
      for (int r = 0; r < 8; ++r) acc[i][j][r] = 0.f;

  const int ar     = lane & 15;
  const int koff8  = (lane >> 4) * 16;
  const int koff16 = (lane >> 4) * 16;

  // staging: 128x32 bf16 tile = 512 chunks of 16B -> 2 chunks/thread/tile
  auto stage = [&](int buf, int k0) {
#pragma unroll
    for (int i = 0; i < 2; ++i) {
      int ci  = tid + i * 256;   // 0..511
      int row = ci >> 2;
      int kc  = (ci & 3) << 3;   // 8 bf16 per chunk
      copy16_g2l(A + (size_t)(m0 + row) * HID + k0 + kc,
                 &As[buf][row * LDK + kc]);
      copy16_g2l(W + (size_t)(n0 + row) * HID + k0 + kc,
                 &Bs[buf][row * LDK + kc]);
#if !USE_ASYNC_LDS
      if (k0 + 32 < HID) {
        __builtin_prefetch(A + (size_t)(m0 + row) * HID + k0 + 32 + kc, 0, 1);
        __builtin_prefetch(W + (size_t)(n0 + row) * HID + k0 + 32 + kc, 0, 1);
      }
#endif
    }
  };

  stage(0, 0);
  async_wait_all();
  __syncthreads();

  for (int k0 = 0; k0 < HID; k0 += 32) {
    const int cur = (k0 >> 5) & 1;
    if (k0 + 32 < HID) stage(cur ^ 1, k0 + 32);  // overlap with WMMA below

    v16bf af[2];
    af[0] = frag_a(&As[cur][(wm * 32 + ar) * LDK + koff8]);
    af[1] = frag_a(&As[cur][(wm * 32 + 16 + ar) * LDK + koff8]);
    v16bf bfr[4];
#pragma unroll
    for (int j = 0; j < 4; ++j)
      bfr[j] = frag_b(&Bs[cur][(wn * 64 + j * 16 + ar) * LDK + koff16]);

#pragma unroll
    for (int i = 0; i < 2; ++i)
#pragma unroll
      for (int j = 0; j < 4; ++j)
        acc[i][j] = __builtin_amdgcn_wmma_f32_16x16x32_bf16(
            false, af[i], false, bfr[j], (short)0, acc[i][j], false, false);

    async_wait_all();
    __syncthreads();
  }

  // epilogue: C layout: lane L, VGPR r -> M = 8*(L>>4)+r, N = L&15
  const int rsub = (lane >> 4) * 8;
#pragma unroll
  for (int i = 0; i < 2; ++i) {
#pragma unroll
    for (int j = 0; j < 4; ++j) {
      const int gn = n0 + wn * 64 + j * 16 + ar;
      const float bv = bias[gn];
#pragma unroll
      for (int r = 0; r < 8; ++r) {
        const int gm = m0 + wm * 32 + i * 16 + rsub + r;
        const float v = (acc[i][j][r] + bv) * scale;
        if (MODE == 0) {
          ((float*)Out)[(size_t)gm * HID + gn] = v;
        } else {
          const int b = gm >> 11, s = gm & (SEQ - 1);
          const int h = gn >> 6,  d = gn & (HDIM - 1);
          ((__bf16*)Out)[((size_t)(b * NHEADS + h) * SEQ + s) * HDIM + d] = (__bf16)v;
        }
      }
    }
  }
}

// ---------------------------------------------------------------------------
// Flash-style attention: one workgroup = 64 queries of one (batch, head).
// Q pre-scaled by 1/sqrt(HDIM). Output ctx bf16 in [B, S, HID] layout.
__global__ __launch_bounds__(256)
void mha_attention(const __bf16* __restrict__ Qb, const __bf16* __restrict__ Kb,
                   const __bf16* __restrict__ Vb, __bf16* __restrict__ Ctx) {
  __shared__ __align__(16) __bf16 Qt[64 * LDA];
  __shared__ __align__(16) __bf16 Kt[64 * LDA];
  __shared__ __align__(16) __bf16 Vt[64 * LDA];  // transposed: [d][key]
  __shared__ __align__(16) float  Sb[64 * LDA];
  __shared__ __align__(16) __bf16 Pb[64 * LDA];
  __shared__ float pmax[256];
  __shared__ float psum[256];
  __shared__ float m_s[64];
  __shared__ float l_s[64];
  __shared__ float alpha_s[64];

  const int tid  = threadIdx.x;
  const int lane = tid & 31;
  const int wid  = tid >> 5;
  const int tm   = wid & 3;   // row tile (16 q-rows)
  const int wn   = wid >> 2;  // col group (32 cols)
  const int q0   = blockIdx.x * 64;
  const int bh   = blockIdx.y;
  const size_t base = (size_t)bh * SEQ * HDIM;

  // stage Q tile once: 64x64 bf16 = 512 16B chunks -> 2/thread
#pragma unroll
  for (int i = 0; i < 2; ++i) {
    int ci  = tid + i * 256;
    int row = ci >> 3;          // 8 chunks per row
    int kc  = (ci & 7) << 3;
    copy16_g2l(Qb + base + (size_t)(q0 + row) * HDIM + kc, &Qt[row * LDA + kc]);
  }
  if (tid < 64) { m_s[tid] = -INFINITY; l_s[tid] = 0.f; }

  v8f acc[2];
#pragma unroll
  for (int r = 0; r < 8; ++r) { acc[0][r] = 0.f; acc[1][r] = 0.f; }

  const int ar     = lane & 15;
  const int rsub   = (lane >> 4) * 8;
  const int koff8  = (lane >> 4) * 16;
  const int koff16 = (lane >> 4) * 16;
  const int srow   = tid >> 2;  // softmax: 4 threads per row
  const int sq     = tid & 3;

  for (int kb = 0; kb < SEQ / 64; ++kb) {
    __syncthreads();  // previous iteration done reading Kt/Vt/Sb/Pb
    // stage K tile (row-major, async) and V tile (transposed, manual)
#pragma unroll
    for (int i = 0; i < 2; ++i) {
      int ci  = tid + i * 256;
      int row = ci >> 3;
      int kc  = (ci & 7) << 3;
      size_t src = base + (size_t)(kb * 64 + row) * HDIM + kc;
      copy16_g2l(Kb + src, &Kt[row * LDA + kc]);
      uint4 vv = *(const uint4*)(Vb + src);
      __bf16 vtmp[8];
      __builtin_memcpy(vtmp, &vv, 16);
#pragma unroll
      for (int e = 0; e < 8; ++e) Vt[(kc + e) * LDA + row] = vtmp[e];
    }
    async_wait_all();
    __syncthreads();

    // S = Q @ K^T  (64x64, each wave 2 tiles)
    v8f st[2];
#pragma unroll
    for (int r = 0; r < 8; ++r) { st[0][r] = 0.f; st[1][r] = 0.f; }
#pragma unroll
    for (int kk = 0; kk < 64; kk += 32) {
      v16bf qf = frag_a(&Qt[(tm * 16 + ar) * LDA + kk + koff8]);
#pragma unroll
      for (int t = 0; t < 2; ++t) {
        v16bf kf = frag_b(&Kt[(wn * 32 + t * 16 + ar) * LDA + kk + koff16]);
        st[t] = __builtin_amdgcn_wmma_f32_16x16x32_bf16(
            false, qf, false, kf, (short)0, st[t], false, false);
      }
    }
#pragma unroll
    for (int t = 0; t < 2; ++t)
#pragma unroll
      for (int r = 0; r < 8; ++r)
        Sb[(tm * 16 + rsub + r) * LDA + wn * 32 + t * 16 + ar] = st[t][r];
    __syncthreads();

    // online softmax, 4 threads per row, 16 cols each
    {
      const float* Sr = &Sb[srow * LDA + sq * 16];
      float mx = -INFINITY;
#pragma unroll
      for (int j = 0; j < 16; ++j) mx = fmaxf(mx, Sr[j]);
      pmax[tid] = mx;
    }
    __syncthreads();
    if (sq == 0) {
      const int t4 = srow * 4;
      float mx = fmaxf(fmaxf(pmax[t4], pmax[t4 + 1]),
                       fmaxf(pmax[t4 + 2], pmax[t4 + 3]));
      mx = fmaxf(mx, m_s[srow]);
      alpha_s[srow] = __expf(m_s[srow] - mx);
      m_s[srow] = mx;
    }
    __syncthreads();
    {
      const float mx = m_s[srow];
      const float* Sr = &Sb[srow * LDA + sq * 16];
      __bf16* Pr = &Pb[srow * LDA + sq * 16];
      float sum = 0.f;
#pragma unroll
      for (int j = 0; j < 16; ++j) {
        const float p = __expf(Sr[j] - mx);
        Pr[j] = (__bf16)p;
        sum += p;
      }
      psum[tid] = sum;
    }
    __syncthreads();
    if (sq == 0) {
      const int t4 = srow * 4;
      l_s[srow] = l_s[srow] * alpha_s[srow] +
                  (psum[t4] + psum[t4 + 1]) + (psum[t4 + 2] + psum[t4 + 3]);
    }

    // rescale running ctx, then ctx += P @ V
#pragma unroll
    for (int r = 0; r < 8; ++r) {
      const float a = alpha_s[tm * 16 + rsub + r];
      acc[0][r] *= a;
      acc[1][r] *= a;
    }
#pragma unroll
    for (int kk = 0; kk < 64; kk += 32) {
      v16bf pf = frag_a(&Pb[(tm * 16 + ar) * LDA + kk + koff8]);
#pragma unroll
      for (int t = 0; t < 2; ++t) {
        v16bf vf = frag_b(&Vt[(wn * 32 + t * 16 + ar) * LDA + kk + koff16]);
        acc[t] = __builtin_amdgcn_wmma_f32_16x16x32_bf16(
            false, pf, false, vf, (short)0, acc[t], false, false);
      }
    }
  }

  __syncthreads();  // final l_s visible

  const int b = bh >> 4, h = bh & 15;
#pragma unroll
  for (int t = 0; t < 2; ++t) {
    const int d = wn * 32 + t * 16 + ar;
#pragma unroll
    for (int r = 0; r < 8; ++r) {
      const int rl = tm * 16 + rsub + r;
      const float v = acc[t][r] / l_s[rl];
      const int s = q0 + rl;
      Ctx[((size_t)(b * SEQ + s)) * HID + h * HDIM + d] = (__bf16)v;
    }
  }
}

// ---------------------------------------------------------------------------
extern "C" void kernel_launch(void* const* d_in, const int* in_sizes, int n_in,
                              void* d_out, int out_size, void* d_ws, size_t ws_size,
                              hipStream_t stream) {
  const float* x  = (const float*)d_in[0];
  const float* Wq = (const float*)d_in[1];
  const float* bq = (const float*)d_in[2];
  const float* Wk = (const float*)d_in[3];
  const float* bk = (const float*)d_in[4];
  const float* Wv = (const float*)d_in[5];
  const float* bv = (const float*)d_in[6];
  const float* Wo = (const float*)d_in[7];
  const float* bo = (const float*)d_in[8];
  float* out = (float*)d_out;

  char* ws = (char*)d_ws;
  const size_t MiB = 1024u * 1024u;
  __bf16* xb  = (__bf16*)(ws + 0 * MiB);   // 8 MiB
  __bf16* wqb = (__bf16*)(ws + 8 * MiB);   // 2 MiB each
  __bf16* wkb = (__bf16*)(ws + 10 * MiB);
  __bf16* wvb = (__bf16*)(ws + 12 * MiB);
  __bf16* wob = (__bf16*)(ws + 14 * MiB);
  __bf16* Qb  = (__bf16*)(ws + 16 * MiB);  // 8 MiB each, [B,heads,S,64]
  __bf16* Kb  = (__bf16*)(ws + 24 * MiB);
  __bf16* Vb  = (__bf16*)(ws + 32 * MiB);
  __bf16* Cb  = (__bf16*)(ws + 40 * MiB);  // ctx bf16 [B,S,HID]

  cvt_f32_to_bf16<<<MTOT * HID / 1024, 256, 0, stream>>>(x, xb, MTOT * HID);
  cvt_f32_to_bf16<<<HID * HID / 1024, 256, 0, stream>>>(Wq, wqb, HID * HID);
  cvt_f32_to_bf16<<<HID * HID / 1024, 256, 0, stream>>>(Wk, wkb, HID * HID);
  cvt_f32_to_bf16<<<HID * HID / 1024, 256, 0, stream>>>(Wv, wvb, HID * HID);
  cvt_f32_to_bf16<<<HID * HID / 1024, 256, 0, stream>>>(Wo, wob, HID * HID);

  dim3 gg(HID / 128, MTOT / 128);
  // Q gets the 1/sqrt(64) score scale folded in
  gemm_bias<1><<<gg, 256, 0, stream>>>(xb, wqb, bq, (void*)Qb, 0.125f);
  gemm_bias<1><<<gg, 256, 0, stream>>>(xb, wkb, bk, (void*)Kb, 1.0f);
  gemm_bias<1><<<gg, 256, 0, stream>>>(xb, wvb, bv, (void*)Vb, 1.0f);

  mha_attention<<<dim3(SEQ / 64, BATCH * NHEADS), 256, 0, stream>>>(Qb, Kb, Vb, Cb);

  gemm_bias<0><<<gg, 256, 0, stream>>>(Cb, wob, bo, (void*)out, 1.0f);

  (void)in_sizes; (void)n_in; (void)out_size; (void)ws_size;
}
